// StateSpaceModel_90117003805110
// MI455X (gfx1250) — matrix operand order repaired and checked by reading, not compiled, not run on hardware
//
#include <hip/hip_runtime.h>
#include <hip/hip_bf16.h>
#include <math.h>

#define D_MODEL 1024
#define D_INNER 2048
#define D_STATE 16
#define D_CONV 4
#define DT_RANK 64
#define N_LAYERS 6
#define BATCH 2
#define SEQ 2048
#define MTOT (BATCH * SEQ) // 4096 rows in every GEMM

// chunked selective scan
#define NCHUNK 32
#define LCHUNK (SEQ / NCHUNK) // 64

typedef __attribute__((ext_vector_type(16))) __bf16 v16bf;
typedef __attribute__((ext_vector_type(8)))  __bf16 v8bf;
typedef __attribute__((ext_vector_type(4)))  __bf16 v4bf;
typedef __attribute__((ext_vector_type(8)))  float  v8f;
typedef __attribute__((ext_vector_type(4)))  int    v4i;

#define LSTR 48  // LDS row stride in bf16 elems (96 B: bank-conflict pad, keeps 16B alignment)

// gfx1250 async global->LDS copy path (ASYNCcnt), guarded so either toolchain compiles
#if __has_builtin(__builtin_amdgcn_global_load_async_to_lds_b128) && \
    __has_builtin(__builtin_amdgcn_s_wait_asynccnt)
#define USE_ASYNC_LDS 1
#else
#define USE_ASYNC_LDS 0
#endif

__device__ __forceinline__ float softplus_f(float x) {
    return (x > 20.f) ? x : log1pf(__expf(x));
}
__device__ __forceinline__ float silu_f(float x) {
    return x / (1.f + __expf(-x));
}

// ---------------------------------------------------------------------------
// f32 -> bf16 weight conversion (n multiple of 4)
// ---------------------------------------------------------------------------
__global__ __launch_bounds__(256)
void cvt_bf16_kernel(const float* __restrict__ in, __bf16* __restrict__ out, int n)
{
    const int i = (blockIdx.x * 256 + threadIdx.x) * 4;
    if (i < n) {
        float4 f = *(const float4*)(in + i);
        v4bf v;
        v[0] = (__bf16)f.x; v[1] = (__bf16)f.y; v[2] = (__bf16)f.z; v[3] = (__bf16)f.w;
        *(v4bf*)(out + i) = v;
    }
}

// ---------------------------------------------------------------------------
// C[M x N] = act(A[M x K] * W[N x K]^T + bias)
// A: f32 row-major (row stride lda), staged to LDS with f32->bf16 convert.
// W: bf16 row-major (N rows of K), staged to LDS with async global->LDS B128
//    when available (falls back to vector load + ds_store).
// Block tile 128(M) x 64(N), K step 32, 256 threads = 8 waves.
// Wave (wm 0..3, wn 0..1): 32x32 output = 4 v_wmma_f32_16x16x32_bf16 / K-step.
// LDS double-buffered. ACT: 0 = none, 1 = softplus.
// ---------------------------------------------------------------------------
template<int ACT, bool HAS_BIAS>
__global__ __launch_bounds__(256)
void gemm_wmma_bf16(const float* __restrict__ A, int lda,
                    const __bf16* __restrict__ W,
                    const float* __restrict__ bias,
                    float* __restrict__ C, int ldc,
                    int K, int N)
{
    __shared__ __bf16 sA[2][128 * LSTR]; // 2 x 12 KB
    __shared__ __bf16 sB[2][64 * LSTR];  // 2 x  6 KB

    const int tid  = threadIdx.x;
    const int lane = tid & 31;
    const int wv   = tid >> 5;
    const int wm   = wv & 3;    // 0..3 : 32-row strip
    const int wn   = wv >> 2;   // 0..1 : 32-col strip
    const int lh   = lane >> 4; // 0/1
    const int ll   = lane & 15;

    const int bm = blockIdx.y * 128;
    const int bn = blockIdx.x * 64;

    // A loader: thread covers row (tid>>1), 16 cols starting at (tid&1)*16
    const int ar = tid >> 1;
    const int ah = tid & 1;
    // W loader: thread covers row (tid>>2), 8 bf16 starting at (tid&3)*8
    const int br = tid >> 2;
    const int bg = tid & 3;

    const float*  aBase = A + (size_t)(bm + ar) * lda + ah * 16;
    const __bf16* wBase = W + (size_t)(bn + br) * K + bg * 8;
    const bool    wOK   = (bn + br) < N;

    v8f acc[2][2] = {};

    const int nk = K >> 5;

    auto stage = [&](int kt, int bsel) {
        // A tile: 128x32 f32 -> bf16
        {
            const float* ap = aBase + kt * 32;
            float4 f0 = *(const float4*)ap;
            float4 f1 = *(const float4*)(ap + 4);
            float4 f2 = *(const float4*)(ap + 8);
            float4 f3 = *(const float4*)(ap + 12);
            v8bf v0, v1;
            v0[0] = (__bf16)f0.x; v0[1] = (__bf16)f0.y; v0[2] = (__bf16)f0.z; v0[3] = (__bf16)f0.w;
            v0[4] = (__bf16)f1.x; v0[5] = (__bf16)f1.y; v0[6] = (__bf16)f1.z; v0[7] = (__bf16)f1.w;
            v1[0] = (__bf16)f2.x; v1[1] = (__bf16)f2.y; v1[2] = (__bf16)f2.z; v1[3] = (__bf16)f2.w;
            v1[4] = (__bf16)f3.x; v1[5] = (__bf16)f3.y; v1[6] = (__bf16)f3.z; v1[7] = (__bf16)f3.w;
            *(v8bf*)&sA[bsel][ar * LSTR + ah * 16]     = v0;
            *(v8bf*)&sA[bsel][ar * LSTR + ah * 16 + 8] = v1;
        }
        // W tile: 64x32 bf16 (16 B per thread)
        __bf16* dst = &sB[bsel][br * LSTR + bg * 8];
#if USE_ASYNC_LDS
        if (wOK) {
            __builtin_amdgcn_global_load_async_to_lds_b128(
                (__attribute__((address_space(1))) v4i*)(wBase + kt * 32),
                (__attribute__((address_space(3))) v4i*)dst,
                0, 0);
        } else {
            v8bf z = {};
            *(v8bf*)dst = z;
        }
#else
        {
            v8bf v = {};
            if (wOK) v = *(const v8bf*)(wBase + kt * 32);
            *(v8bf*)dst = v;
        }
#endif
    };

    stage(0, 0);

    for (int kt = 0; kt < nk; ++kt) {
#if USE_ASYNC_LDS
        __builtin_amdgcn_s_wait_asynccnt(0); // W tile of buf[kt&1] landed in LDS
#endif
        __syncthreads(); // buf[kt&1] ready; reads of buf[(kt+1)&1] from kt-1 done

        if (kt + 1 < nk) {
            stage(kt + 1, (kt + 1) & 1);
            if (kt + 2 < nk) { // prefetch tile kt+2 (global_prefetch_b8)
                __builtin_prefetch(aBase + (kt + 2) * 32, 0, 1);
                if (wOK) __builtin_prefetch(wBase + (kt + 2) * 32, 0, 1);
            }
        }

        const int cb = kt & 1;

        // A fragments (ISA 16-bit A 16x32 layout):
        // lanes 0-15: row=ll, K=0..7 & 16..23 ; lanes 16-31: K=8..15 & 24..31
        v16bf af[2];
        #pragma unroll
        for (int am = 0; am < 2; ++am) {
            const __bf16* p = &sA[cb][(wm * 32 + am * 16 + ll) * LSTR + lh * 8];
            v8bf a0 = *(const v8bf*)p;
            v8bf a1 = *(const v8bf*)(p + 16);
            #pragma unroll
            for (int i = 0; i < 8; ++i) { af[am][i] = a0[i]; af[am][8 + i] = a1[i]; }
        }
        // B fragments (B 32x16: lane = col, lanes 0-15 K=0..7 & 16..23, 16-31 K=8..15 & 24..31)
        v16bf bfv[2];
        #pragma unroll
        for (int t = 0; t < 2; ++t) {
            const __bf16* p = &sB[cb][(wn * 32 + t * 16 + ll) * LSTR + lh * 8];
            v8bf b0 = *(const v8bf*)p;
            v8bf b1 = *(const v8bf*)(p + 16);
            #pragma unroll
            for (int i = 0; i < 8; ++i) { bfv[t][i] = b0[i]; bfv[t][8 + i] = b1[i]; }
        }

        #pragma unroll
        for (int am = 0; am < 2; ++am)
            #pragma unroll
            for (int t = 0; t < 2; ++t)
                acc[am][t] = __builtin_amdgcn_wmma_f32_16x16x32_bf16(
                    false, af[am], false, bfv[t], (short)0, acc[am][t], false, false);
    }

    // epilogue: C/D layout -> element (vgpr rr, lane) = (M = lh*8+rr, N = ll)
    #pragma unroll
    for (int am = 0; am < 2; ++am) {
        #pragma unroll
        for (int t = 0; t < 2; ++t) {
            const int col = bn + wn * 32 + t * 16 + ll;
            if (col < N) {
                float bv = HAS_BIAS ? bias[col] : 0.f;
                #pragma unroll
                for (int rr = 0; rr < 8; ++rr) {
                    const int row = bm + wm * 32 + am * 16 + lh * 8 + rr;
                    float v = acc[am][t][rr] + bv;
                    if (ACT == 1) v = softplus_f(v);
                    C[(size_t)row * ldc + col] = v;
                }
            }
        }
    }
}

// ---------------------------------------------------------------------------
// Depthwise causal conv (width 4) over seq + bias + SiLU.
// ---------------------------------------------------------------------------
__global__ __launch_bounds__(256)
void conv_silu_kernel(const float* __restrict__ xz,
                      const float* __restrict__ cw,
                      const float* __restrict__ cb,
                      float* __restrict__ xr)
{
    const int idx = blockIdx.x * 256 + threadIdx.x;
    const int c = idx & (D_INNER - 1);
    const int t = (idx / D_INNER) & (SEQ - 1);
    const int b = idx / (D_INNER * SEQ);
    const float* w = cw + c * D_CONV;
    float s = cb[c];
    #pragma unroll
    for (int k = 0; k < D_CONV; ++k) {
        const int tt = t - (D_CONV - 1) + k;
        if (tt >= 0)
            s += w[k] * xz[((size_t)b * SEQ + tt) * (2 * D_INNER) + c];
    }
    xr[idx] = silu_f(s);
}

// ---------------------------------------------------------------------------
// Chunked selective scan.
// The per-(d,n) recurrence h_t = a_t h_{t-1} + b_t is associative; split SEQ
// into NCHUNK chunks: pass1 = per-chunk decay product P and end state (h from
// zero), pass2 = sequential composition over the 32 chunk boundaries giving
// each chunk's true start state, pass3 = replay chunks from the correct start
// with the fused D*u skip + silu(z) gating epilogue (y overwrites xr).
// State layout: [b][chunk][n][d] so d-consecutive threads stay coalesced.
// ---------------------------------------------------------------------------
__global__ __launch_bounds__(256)
void scan_pass1(const float* __restrict__ xr,
                const float* __restrict__ dt,
                const float* __restrict__ xdbl,
                const float* __restrict__ a_log,
                float* __restrict__ Pbuf,
                float* __restrict__ Hbuf)
{
    const int dg = blockIdx.x & (D_INNER / 256 - 1);       // 0..7
    const int ch = (blockIdx.x >> 3) & (NCHUNK - 1);       // 0..31
    const int b  = blockIdx.x >> 8;
    const int d  = dg * 256 + threadIdx.x;

    float An[D_STATE];
    #pragma unroll
    for (int n = 0; n < D_STATE; ++n)
        An[n] = -__expf(a_log[d * D_STATE + n]);

    float P[D_STATE];
    float h[D_STATE];
    #pragma unroll
    for (int n = 0; n < D_STATE; ++n) { P[n] = 1.f; h[n] = 0.f; }

    const int t0 = ch * LCHUNK;
    for (int t = t0; t < t0 + LCHUNK; ++t) {
        const size_t row = (size_t)b * SEQ + t;
        const float u   = xr[row * D_INNER + d];
        const float dtv = dt[row * D_INNER + d];
        const float* xd = xdbl + row * (DT_RANK + 2 * D_STATE);
        const float du = dtv * u;
        #pragma unroll
        for (int n = 0; n < D_STATE; ++n) {
            const float e = __expf(dtv * An[n]);
            P[n] *= e;
            h[n] = e * h[n] + du * xd[DT_RANK + n];
        }
    }
    const size_t base = (((size_t)b * NCHUNK + ch) * D_STATE) * D_INNER + d;
    #pragma unroll
    for (int n = 0; n < D_STATE; ++n) {
        Pbuf[base + (size_t)n * D_INNER] = P[n];
        Hbuf[base + (size_t)n * D_INNER] = h[n];
    }
}

__global__ __launch_bounds__(256)
void scan_compose(const float* __restrict__ Pbuf, float* __restrict__ Hbuf)
{
    // thread per (b, n, d): composes chunk boundaries; Hbuf becomes h_start.
    const int idx = blockIdx.x * 256 + threadIdx.x;
    const int d = idx & (D_INNER - 1);
    const int n = (idx >> 11) & (D_STATE - 1);
    const int b = idx >> 15;
    float prior = 0.f;
    for (int c = 0; c < NCHUNK; ++c) {
        const size_t o = (((size_t)b * NCHUNK + c) * D_STATE + n) * D_INNER + d;
        const float Pv = Pbuf[o];
        const float hv = Hbuf[o];
        Hbuf[o] = prior;            // true start state of chunk c
        prior = Pv * prior + hv;
    }
}

__global__ __launch_bounds__(256)
void scan_pass3(float* __restrict__ xr,          // in: u, out: gated y
                const float* __restrict__ dt,
                const float* __restrict__ xdbl,
                const float* __restrict__ xz,    // z = xz[..., D_INNER:]
                const float* __restrict__ a_log,
                const float* __restrict__ Dp,
                const float* __restrict__ Hstart)
{
    const int dg = blockIdx.x & (D_INNER / 256 - 1);
    const int ch = (blockIdx.x >> 3) & (NCHUNK - 1);
    const int b  = blockIdx.x >> 8;
    const int d  = dg * 256 + threadIdx.x;

    float An[D_STATE];
    #pragma unroll
    for (int n = 0; n < D_STATE; ++n)
        An[n] = -__expf(a_log[d * D_STATE + n]);
    const float Dd = Dp[d];

    float h[D_STATE];
    const size_t base = (((size_t)b * NCHUNK + ch) * D_STATE) * D_INNER + d;
    #pragma unroll
    for (int n = 0; n < D_STATE; ++n)
        h[n] = Hstart[base + (size_t)n * D_INNER];

    const int t0 = ch * LCHUNK;
    for (int t = t0; t < t0 + LCHUNK; ++t) {
        const size_t row = (size_t)b * SEQ + t;
        const float u   = xr[row * D_INNER + d];
        const float dtv = dt[row * D_INNER + d];
        const float zv  = xz[row * (2 * D_INNER) + D_INNER + d];
        const float* xd = xdbl + row * (DT_RANK + 2 * D_STATE);
        const float du = dtv * u;
        float y = 0.f;
        #pragma unroll
        for (int n = 0; n < D_STATE; ++n) {
            h[n] = __expf(dtv * An[n]) * h[n] + du * xd[DT_RANK + n];
            y += h[n] * xd[DT_RANK + D_STATE + n];
        }
        xr[row * D_INNER + d] = (y + Dd * u) * silu_f(zv);
    }
}

// ---------------------------------------------------------------------------
// In-place LayerNorm over rows of D_MODEL. One block (256 thr) per row.
// ---------------------------------------------------------------------------
__global__ __launch_bounds__(256)
void layernorm_kernel(float* __restrict__ x,
                      const float* __restrict__ w,
                      const float* __restrict__ b)
{
    __shared__ float red[256];
    const int tid = threadIdx.x;
    float* xp = x + (size_t)blockIdx.x * D_MODEL;

    float v[D_MODEL / 256];
    float s = 0.f;
    #pragma unroll
    for (int i = 0; i < D_MODEL / 256; ++i) { v[i] = xp[tid + i * 256]; s += v[i]; }
    red[tid] = s; __syncthreads();
    #pragma unroll
    for (int o = 128; o > 0; o >>= 1) { if (tid < o) red[tid] += red[tid + o]; __syncthreads(); }
    const float mean = red[0] * (1.f / D_MODEL);
    __syncthreads();

    float q = 0.f;
    #pragma unroll
    for (int i = 0; i < D_MODEL / 256; ++i) { const float dd = v[i] - mean; q += dd * dd; }
    red[tid] = q; __syncthreads();
    #pragma unroll
    for (int o = 128; o > 0; o >>= 1) { if (tid < o) red[tid] += red[tid + o]; __syncthreads(); }
    const float rstd = rsqrtf(red[0] * (1.f / D_MODEL) + 1e-5f);

    #pragma unroll
    for (int i = 0; i < D_MODEL / 256; ++i) {
        const int c = tid + i * 256;
        xp[c] = (v[i] - mean) * rstd * w[c] + b[c];
    }
}

// ---------------------------------------------------------------------------
// Decoder head: out[row] = sigmoid(dot(x[row,:], dec_w) + dec_b).
// ---------------------------------------------------------------------------
__global__ __launch_bounds__(256)
void dec_kernel(const float* __restrict__ x,
                const float* __restrict__ w,
                const float* __restrict__ bb,
                float* __restrict__ out)
{
    const int lane = threadIdx.x & 31;
    const int wid  = threadIdx.x >> 5;
    const int row  = blockIdx.x * 8 + wid;
    const float* xp = x + (size_t)row * D_MODEL;
    float s = 0.f;
    for (int j = 0; j < D_MODEL; j += 32)
        s += xp[j + lane] * w[j + lane];
    #pragma unroll
    for (int m = 16; m > 0; m >>= 1)
        s += __shfl_xor(s, m, 32);
    if (lane == 0)
        out[row] = 1.f / (1.f + __expf(-(s + bb[0])));
}

// ---------------------------------------------------------------------------
extern "C" void kernel_launch(void* const* d_in, const int* in_sizes, int n_in,
                              void* d_out, int out_size, void* d_ws, size_t ws_size,
                              hipStream_t stream)
{
    const float* src  = (const float*)d_in[0];
    const float* ipw  = (const float*)d_in[1];
    const float* cw   = (const float*)d_in[2];
    const float* cb   = (const float*)d_in[3];
    const float* xpw  = (const float*)d_in[4];
    const float* dpw  = (const float*)d_in[5];
    const float* dpb  = (const float*)d_in[6];
    const float* alog = (const float*)d_in[7];
    const float* Dp   = (const float*)d_in[8];
    const float* opw  = (const float*)d_in[9];
    const float* lnw  = (const float*)d_in[10];
    const float* lnb  = (const float*)d_in[11];
    const float* dw   = (const float*)d_in[12];
    const float* db   = (const float*)d_in[13];
    float* out = (float*)d_out;

    // workspace carve-up (all 16B aligned)
    char* ws = (char*)d_ws;
    float* xz   = (float*)ws; ws += (size_t)MTOT * 2 * D_INNER * sizeof(float); // 64 MB
    float* xr   = (float*)ws; ws += (size_t)MTOT * D_INNER * sizeof(float);     // 32 MB
    float* xdbl = (float*)ws; ws += (size_t)MTOT * 96 * sizeof(float);          // 1.5 MB
    float* dtb  = (float*)ws; ws += (size_t)MTOT * D_INNER * sizeof(float);     // 32 MB
    float* xA   = (float*)ws; ws += (size_t)MTOT * D_MODEL * sizeof(float);     // 16 MB
    float* xB   = (float*)ws; ws += (size_t)MTOT * D_MODEL * sizeof(float);     // 16 MB
    const size_t stN = (size_t)BATCH * NCHUNK * D_STATE * D_INNER;              // 2M elems
    float* Pbuf = (float*)ws; ws += stN * sizeof(float);                        // 8 MB
    float* Hbuf = (float*)ws; ws += stN * sizeof(float);                        // 8 MB
    // per-layer bf16 weight staging
    __bf16* ipw_bf = (__bf16*)ws; ws += (size_t)2 * D_INNER * D_MODEL * sizeof(__bf16);
    __bf16* xpw_bf = (__bf16*)ws; ws += (size_t)96 * D_INNER * sizeof(__bf16);
    __bf16* dpw_bf = (__bf16*)ws; ws += (size_t)D_INNER * DT_RANK * sizeof(__bf16);
    __bf16* opw_bf = (__bf16*)ws; ws += (size_t)D_MODEL * D_INNER * sizeof(__bf16);

    const float* cur = src;
    float* pp[2] = { xA, xB };
    dim3 blk(256);

    for (int l = 0; l < N_LAYERS; ++l) {
        float* nxt = pp[l & 1];

        // 0) convert this layer's weights to bf16 staging
        {
            const int n1 = 2 * D_INNER * D_MODEL;  // 4.19M
            const int n2 = 96 * D_INNER;           // 196K
            const int n3 = D_INNER * DT_RANK;      // 131K
            const int n4 = D_MODEL * D_INNER;      // 2.10M
            cvt_bf16_kernel<<<n1 / 1024, blk, 0, stream>>>(ipw + (size_t)l * n1, ipw_bf, n1);
            cvt_bf16_kernel<<<n2 / 1024, blk, 0, stream>>>(xpw + (size_t)l * n2, xpw_bf, n2);
            cvt_bf16_kernel<<<n3 / 1024, blk, 0, stream>>>(dpw + (size_t)l * n3, dpw_bf, n3);
            cvt_bf16_kernel<<<n4 / 1024, blk, 0, stream>>>(opw + (size_t)l * n4, opw_bf, n4);
        }

        // 1) in_proj: (4096x1024) @ (4096x1024)^T -> xz (4096x4096)
        gemm_wmma_bf16<0, false><<<dim3((2 * D_INNER) / 64, MTOT / 128), blk, 0, stream>>>(
            cur, D_MODEL, ipw_bf, nullptr, xz, 2 * D_INNER, D_MODEL, 2 * D_INNER);

        // 2) depthwise conv + SiLU -> xr
        conv_silu_kernel<<<(MTOT * D_INNER) / 256, blk, 0, stream>>>(
            xz, cw + (size_t)l * D_INNER * D_CONV, cb + (size_t)l * D_INNER, xr);

        // 3) x_proj: (4096x2048) @ (96x2048)^T -> xdbl (4096x96)
        gemm_wmma_bf16<0, false><<<dim3(2, MTOT / 128), blk, 0, stream>>>(
            xr, D_INNER, xpw_bf, nullptr, xdbl, 96, D_INNER, 96);

        // 4) dt_proj + bias + softplus: (4096x64, lda=96) @ (2048x64)^T -> dtb
        gemm_wmma_bf16<1, true><<<dim3(D_INNER / 64, MTOT / 128), blk, 0, stream>>>(
            xdbl, 96, dpw_bf, dpb + (size_t)l * D_INNER, dtb, D_INNER, DT_RANK, D_INNER);

        // 5) chunked selective scan (fused D*u skip + silu(z) gate), y -> xr
        const float* al = alog + (size_t)l * D_INNER * D_STATE;
        scan_pass1<<<BATCH * NCHUNK * (D_INNER / 256), blk, 0, stream>>>(
            xr, dtb, xdbl, al, Pbuf, Hbuf);
        scan_compose<<<(BATCH * D_STATE * D_INNER) / 256, blk, 0, stream>>>(Pbuf, Hbuf);
        scan_pass3<<<BATCH * NCHUNK * (D_INNER / 256), blk, 0, stream>>>(
            xr, dtb, xdbl, xz, al, Dp + (size_t)l * D_INNER, Hbuf);

        // 6) out_proj: (4096x2048) @ (1024x2048)^T -> nxt (4096x1024)
        gemm_wmma_bf16<0, false><<<dim3(D_MODEL / 64, MTOT / 128), blk, 0, stream>>>(
            xr, D_INNER, opw_bf, nullptr, nxt, D_MODEL, D_INNER, D_MODEL);

        // 7) LayerNorm in place
        layernorm_kernel<<<MTOT, blk, 0, stream>>>(
            nxt, lnw + (size_t)l * D_MODEL, lnb + (size_t)l * D_MODEL);

        cur = nxt;
    }

    // decoder head -> sigmoid, out (B*S,)
    dec_kernel<<<MTOT / 8, blk, 0, stream>>>(cur, dw, db, out);
}
